// SwitchedConvHardRouting_28561532518442
// MI455X (gfx1250) — matrix operand order repaired and checked by reading, not compiled, not run on hardware
//
#include <hip/hip_runtime.h>
#include <stddef.h>

// Problem constants (from reference): B, CIN, COUT, S, K, H, W = 8,128,128,8,3,128,128
#define B_      8
#define CIN_    128
#define COUT_   128
#define S_      8
#define KSZ_    3
#define H_      128
#define W_      128
#define HW_     (H_ * W_)            // 16384
#define KDIM_   (CIN_ * KSZ_ * KSZ_) // 1152
#define KCH_    (KDIM_ / 32)         // 36 K-chunks of 32 (one conv tap = 4 chunks)
#define NT_     (COUT_ / 16)         // 8 N-tiles of 16
#define FRAG_   512                  // one B fragment: 32 lanes * 16 bf16 (1024 B)
#define BLDS_BYTES (KCH_ * NT_ * FRAG_ * 2)   // 294,912 B: one branch's B in LDS

typedef __attribute__((ext_vector_type(16))) __bf16 v16bf;
typedef __attribute__((ext_vector_type(8)))  float  v8f;
typedef __attribute__((ext_vector_type(4)))  unsigned int v4u;

__device__ __forceinline__ unsigned short f32_to_bf16_raw(float f) {
  union { float f; unsigned u; } v; v.f = f;
  unsigned u = v.u + 0x7FFFu + ((v.u >> 16) & 1u);  // round-to-nearest-even
  return (unsigned short)(u >> 16);
}

__device__ __forceinline__ v16bf frag_from_u32x8(v4u lo, v4u hi) {
  union { v4u u[2]; v16bf b; } t;
  t.u[0] = lo; t.u[1] = hi;
  return t.b;
}

// ---------------------------------------------------------------------------
// 1) Router: per-pixel argmax over S (softmax is monotone -> argmax(selector)),
//    compact pixel ids into per-(b,s) buckets via atomics. Output deterministic:
//    each pixel's result is independent of bucket ordering.
// ---------------------------------------------------------------------------
__global__ __launch_bounds__(256) void route_kernel(
    const float* __restrict__ sel, int* __restrict__ counts,
    int* __restrict__ buckets) {
  int pid = blockIdx.x * 256 + threadIdx.x;      // 0 .. B*HW-1
  if (pid >= B_ * HW_) return;
  int b = pid / HW_;
  int p = pid - b * HW_;
  const float* sp = sel + (size_t)b * S_ * HW_ + p;
  float best = sp[0];
  int bi = 0;
#pragma unroll
  for (int s = 1; s < S_; ++s) {
    float v = sp[(size_t)s * HW_];
    if (v > best) { best = v; bi = s; }          // strict '>' keeps first max
  }
  int pos = atomicAdd(&counts[b * S_ + bi], 1);
  buckets[(size_t)(b * S_ + bi) * HW_ + pos] = p;
}

// ---------------------------------------------------------------------------
// 2) Input transpose: f32 NCHW -> bf16 NHWC via LDS tile (32 pixels x 128 ch).
//    Reads coalesced over pixels, writes coalesced over channels (packed u32).
// ---------------------------------------------------------------------------
__global__ __launch_bounds__(256) void nhwc_kernel(
    const float* __restrict__ in, unsigned short* __restrict__ outb) {
  __shared__ unsigned short lds[32][CIN_ + 2];   // 8320 B, pad -> conflict-free
  const int b  = blockIdx.y;
  const int p0 = blockIdx.x * 32;
  const int t  = threadIdx.x;
  {
    const int pl = t & 31;
    const int cl = t >> 5;                       // 0..7
    const float* ib = in + (size_t)b * CIN_ * HW_ + p0 + pl;
#pragma unroll
    for (int it = 0; it < CIN_ / 8; ++it) {
      int ci = cl + it * 8;
      lds[pl][ci] = f32_to_bf16_raw(ib[(size_t)ci * HW_]);
    }
  }
  __syncthreads();
  {
    const int c2  = t & 63;                      // u32 column 0..63
    const int pl2 = t >> 6;                      // 0..3
    unsigned* outw = (unsigned*)outb + ((size_t)b * HW_ + p0) * (CIN_ / 2);
#pragma unroll
    for (int it = 0; it < 8; ++it) {
      int p = pl2 + it * 4;
      outw[(size_t)p * (CIN_ / 2) + c2] = *(const unsigned*)&lds[p][2 * c2];
    }
  }
}

// ---------------------------------------------------------------------------
// 3) Weight prep: f32 -> bf16, pre-swizzled into WMMA B-fragment order with a
//    split-half intra-fragment layout for conflict-free LDS reads:
//      elems 0..7  of lane L -> ushort offset  L*8 + e        (bytes [0,512))
//      elems 8..15 of lane L -> ushort offset  256 + L*8 + e-8 (bytes [512,1024))
//    B logical layout (32x16, 16-bit): n = lane%16, h = lane/16;
//    K = 8h + e for e < 8, K = 16 + 8h + (e-8) for e >= 8.
//    K ordering: k = (kh*3+kw)*CIN + ci (tap-major, one tap = 4 32-chunks).
// ---------------------------------------------------------------------------
__global__ __launch_bounds__(256) void wprep_kernel(
    const float* __restrict__ w, unsigned short* __restrict__ wb) {
  const int total = S_ * KCH_ * NT_ * FRAG_;     // 1,179,648
  int idx = blockIdx.x * 256 + threadIdx.x;
  if (idx >= total) return;
  int e    = idx & 15;
  int lane = (idx >> 4) & 31;
  int f    = idx >> 9;
  int nt = f & (NT_ - 1);
  int kc = (f >> 3) % KCH_;
  int s  = f / (KCH_ * NT_);
  int nl = lane & 15;
  int kl = e + ((lane & 16) ? 8 : 0) + ((e & 8) ? 8 : 0);
  int k  = kc * 32 + kl;
  int tap = k >> 7;                              // k / CIN_
  int ci  = k & (CIN_ - 1);
  int cout = nt * 16 + nl;
  // split-half position inside the 512-ushort fragment
  int pos = (e < 8) ? (lane * 8 + e) : (256 + lane * 8 + (e - 8));
  // weight layout: [COUT][CIN][S][3][3]
  float v = w[(((size_t)cout * CIN_ + ci) * S_ + s) * (KSZ_ * KSZ_) + tap];
  wb[(size_t)f * FRAG_ + pos] = f32_to_bf16_raw(v);
}

// ---------------------------------------------------------------------------
// 4) Bucketed im2col GEMM, bf16 WMMA, B staged in 288 KB LDS.
//    Block: 256 thr = 8 waves, tile = 128 pixels x 128 couts.
//    Wave (wm 0..3, wn 0..1): 32 pixels x 64 couts -> 8 accumulators.
//    Per K-chunk: 4 global b128 (A) + 8 ds_load_b128 (B) -> 8 WMMAs.
// ---------------------------------------------------------------------------
__global__ __launch_bounds__(256) void sconv_wmma_kernel(
    const unsigned short* __restrict__ nhwc, const float* __restrict__ bias,
    const int* __restrict__ counts, const int* __restrict__ buckets,
    const unsigned short* __restrict__ wb, float* __restrict__ out) {
  extern __shared__ char smem_raw[];             // BLDS_BYTES, dynamic
  const int b = blockIdx.z;
  const int s = blockIdx.y;
  const int count = counts[b * S_ + s];
  const int tileBase = blockIdx.x * 128;
  if (tileBase >= count) return;                 // block-uniform exit

  // ---- stage this branch's swizzled B into LDS (linear b128 copy) ----
  {
    const v4u* srcB = (const v4u*)(wb + (size_t)s * KCH_ * NT_ * FRAG_);
    v4u* dstB = (v4u*)smem_raw;
    for (int i = threadIdx.x; i < BLDS_BYTES / 16; i += 256) dstB[i] = srcB[i];
  }
  __syncthreads();

  const int* bucket = buckets + (size_t)(b * S_ + s) * HW_;
  const int wave = threadIdx.x >> 5;
  const int lane = threadIdx.x & 31;
  const int wm = wave >> 1;                      // 0..3 (32-pixel group)
  const int wn = wave & 1;                       // 0..1 (64-cout half)
  const int laneHi = (lane >> 4) & 1;
  const int lml = lane & 15;
  const int mBase = tileBase + wm * 32;

  // two A-row pixels per lane (frag rows mBase+lml and mBase+16+lml)
  int r0 = mBase + lml;
  int r1 = r0 + 16;
  int pix0 = bucket[(r0 < count) ? r0 : (count - 1)];
  int pix1 = bucket[(r1 < count) ? r1 : (count - 1)];
  int ph0 = pix0 >> 7, pw0 = pix0 & (W_ - 1);    // W_ == 128
  int ph1 = pix1 >> 7, pw1 = pix1 & (W_ - 1);
  const unsigned short* inB = nhwc + (size_t)b * HW_ * CIN_;

  v8f acc[2][4] = {};
  const v4u vzero = {0u, 0u, 0u, 0u};
  const char* ldsB = smem_raw;
  const int laneOff = lane * 16;                 // bytes within a 512 B half

  for (int tap = 0; tap < KSZ_ * KSZ_; ++tap) {
    int dh = tap / 3 - 1;
    int dw = tap - (tap / 3) * 3 - 1;
    int hh0 = ph0 + dh, ww0 = pw0 + dw;
    int hh1 = ph1 + dh, ww1 = pw1 + dw;
    bool ok0 = ((unsigned)hh0 < (unsigned)H_) && ((unsigned)ww0 < (unsigned)W_);
    bool ok1 = ((unsigned)hh1 < (unsigned)H_) && ((unsigned)ww1 < (unsigned)W_);
    const unsigned short* tp0 = inB + ((size_t)hh0 * W_ + ww0) * CIN_ + laneHi * 8;
    const unsigned short* tp1 = inB + ((size_t)hh1 * W_ + ww1) * CIN_ + laneHi * 8;

#pragma unroll
    for (int c4 = 0; c4 < 4; ++c4) {
      const int kc  = tap * 4 + c4;
      const int cib = c4 << 5;
      // A fragments: two aligned b128 per pixel row group
      v4u a0lo = ok0 ? *(const v4u*)(tp0 + cib)      : vzero;
      v4u a0hi = ok0 ? *(const v4u*)(tp0 + cib + 16) : vzero;
      v4u a1lo = ok1 ? *(const v4u*)(tp1 + cib)      : vzero;
      v4u a1hi = ok1 ? *(const v4u*)(tp1 + cib + 16) : vzero;
      v16bf a0 = frag_from_u32x8(a0lo, a0hi);
      v16bf a1 = frag_from_u32x8(a1lo, a1hi);

#pragma unroll
      for (int j = 0; j < 4; ++j) {
        const int fo = (kc * NT_ + wn * 4 + j) * (FRAG_ * 2);  // fragment byte base
        v4u blo = *(const v4u*)(ldsB + fo + laneOff);          // ds_load_b128
        v4u bhi = *(const v4u*)(ldsB + fo + 512 + laneOff);    // ds_load_b128
        v16bf bf = frag_from_u32x8(blo, bhi);
        acc[0][j] = __builtin_amdgcn_wmma_f32_16x16x32_bf16(
            false, a0, false, bf, (short)0, acc[0][j], false, false);
        acc[1][j] = __builtin_amdgcn_wmma_f32_16x16x32_bf16(
            false, a1, false, bf, (short)0, acc[1][j], false, false);
      }
    }
  }

  // Epilogue: C/D layout -> lane holds n = lml, rows m = jj + 8*laneHi; fuse bias.
#pragma unroll
  for (int j = 0; j < 4; ++j) {
    const int cout = wn * 64 + j * 16 + lml;
    const float bv = bias[cout];
    float* outp = out + ((size_t)b * COUT_ + cout) * HW_;
#pragma unroll
    for (int i = 0; i < 2; ++i) {
#pragma unroll
      for (int jj = 0; jj < 8; ++jj) {
        int m = jj + (laneHi ? 8 : 0) + i * 16;
        int rr = mBase + m;
        if (rr < count) {
          outp[bucket[rr]] = acc[i][j][jj] + bv;
        }
      }
    }
  }
}

// ---------------------------------------------------------------------------
extern "C" void kernel_launch(void* const* d_in, const int* in_sizes, int n_in,
                              void* d_out, int out_size, void* d_ws, size_t ws_size,
                              hipStream_t stream) {
  (void)in_sizes; (void)n_in; (void)out_size; (void)ws_size;
  const float* input    = (const float*)d_in[0];  // [B,CIN,H,W]
  const float* selector = (const float*)d_in[1];  // [B,S,H,W]
  const float* weight   = (const float*)d_in[2];  // [COUT,CIN,S,3,3]
  const float* bias     = (const float*)d_in[3];  // [COUT]
  float* out = (float*)d_out;                     // [B,COUT,H,W]

  // Workspace layout
  char* ws = (char*)d_ws;
  int* counts  = (int*)ws;                                   // 64 ints (256 B slot)
  int* buckets = (int*)(ws + 256);                           // B*S*HW ints = 4 MiB
  size_t off_wb   = 256 + (size_t)B_ * S_ * HW_ * sizeof(int);
  size_t off_nhwc = off_wb + (size_t)S_ * KCH_ * NT_ * FRAG_ * sizeof(unsigned short);
  unsigned short* wbuf  = (unsigned short*)(ws + off_wb);    // 2.25 MiB bf16 weights
  unsigned short* nhwc  = (unsigned short*)(ws + off_nhwc);  // 32 MiB bf16 NHWC input

  hipMemsetAsync(counts, 0, 256, stream);

  route_kernel<<<(B_ * HW_ + 255) / 256, 256, 0, stream>>>(selector, counts, buckets);

  nhwc_kernel<<<dim3(HW_ / 32, B_), 256, 0, stream>>>(input, nhwc);

  const int wtotal = S_ * KCH_ * NT_ * FRAG_;
  wprep_kernel<<<(wtotal + 255) / 256, 256, 0, stream>>>(weight, wbuf);

  // 288 KB dynamic LDS per workgroup (CDNA5 allows up to 320 KB)
  sconv_wmma_kernel<<<dim3(HW_ / 128, S_, B_), 256, BLDS_BYTES, stream>>>(
      nhwc, bias, counts, buckets, wbuf, out);
}
